// AuTx2D_44032004719304
// MI455X (gfx1250) — matrix operand order repaired and checked by reading, not compiled, not run on hardware
//
#include <hip/hip_runtime.h>

// ---------------------------------------------------------------------------
// AuTx2D forward on gfx1250 (MI455X): WMMA f16 GEMMs + wave32 VALU attention/LN
//  - B weights pre-packed to fragment-native layout (k/32, n, k%32)
//  - 128x128 block tile, 32x64 wave tile (8 WMMA / K-step), double-buffered LDS
// ---------------------------------------------------------------------------

typedef _Float16 half_t;
typedef __attribute__((ext_vector_type(16))) _Float16 v16h;
typedef __attribute__((ext_vector_type(8)))  float    v8f;

union HFrag { v16h v; half_t h[16]; };

#define D_MODEL 768
#define D_FF    3072
#define NB      256        // batch
#define SEQ     65         // 1 + 64 tokens
#define NTOK    64
#define MROWS   (NB * SEQ)     // 16640
#define MTOKS   (NB * NTOK)    // 16384
#define NHEAD   12
#define DKV     64

__device__ __forceinline__ float wave_sum(float v) {
#pragma unroll
  for (int o = 16; o > 0; o >>= 1) v += __shfl_xor(v, o, 32);
  return v;
}
__device__ __forceinline__ float wave_max(float v) {
#pragma unroll
  for (int o = 16; o > 0; o >>= 1) v = fmaxf(v, __shfl_xor(v, o, 32));
  return v;
}
__device__ __forceinline__ float gelu_exact(float x) {
  return 0.5f * x * (1.0f + erff(x * 0.70710678118654752f));
}

// ---------- weight f32 -> f16 with fragment-native packing -----------------
// src row-major [K,N]; dst: Bp[(k>>5)*N*32 + n*32 + (k&31)]
__global__ void cvt_pack_b(const float* __restrict__ src, half_t* __restrict__ dst,
                           int K, int N) {
  int i = blockIdx.x * 256 + threadIdx.x;
  if (i >= K * N) return;
  int k = i / N, n = i - k * N;
  dst[((size_t)(k >> 5) * N + n) * 32 + (k & 31)] = (half_t)src[i];
}

// ------------------------- patchify gather ---------------------------------
// P[b*64 + tt*4 + ss][p*16 + q] = inputs[b, tt*4+p, ss*16+q, 0]
__global__ void patchify(const float* __restrict__ in, float* __restrict__ P) {
  int i = blockIdx.x * 256 + threadIdx.x;      // MTOKS*64 = 1048576 exact
  int c = i & 63, r = i >> 6;
  int b = r >> 6, tok = r & 63;
  int tt = tok >> 2, ss = tok & 3;
  int p = c >> 4, q = c & 15;
  P[i] = in[(((size_t)b * 64) + tt * 4 + p) * 64 + ss * 16 + q];
}

// ------------------------- AGG token fill ----------------------------------
__global__ void set_agg(const float* __restrict__ agg, float* __restrict__ H) {
  int i = blockIdx.x * 256 + threadIdx.x;      // NB*768 = 196608 exact
  int b = i / D_MODEL, d = i - b * D_MODEL;
  H[((size_t)b * SEQ) * D_MODEL + d] = agg[d];
}

// ------------- WMMA GEMM: C = A(f32->f16) @ Bpacked(f16) + bias ------------
// A row-major [M,K]; Bp packed (see cvt_pack_b); C row-major [M,N]
// Block tile 128x128, wave tile 32x64, BK=32, double-buffered LDS.
#define BM 128
#define BN 128
#define BK 32
#define LS 40   // padded LDS row stride in halves (80B, 16B-aligned rows)

__device__ __forceinline__ v8f wmma16(const HFrag& a, const HFrag& b, v8f c) {
  return __builtin_amdgcn_wmma_f32_16x16x32_f16(false, a.v, false, b.v,
                                                (short)0, c, false, false);
}

__global__ __launch_bounds__(256) void gemm_wmma(
    const float* __restrict__ A, const half_t* __restrict__ Bp,
    float* __restrict__ C, const float* __restrict__ bias,
    int M, int N, int K, int do_gelu)
{
  __shared__ half_t As[2][BM * LS];   // [row][k]
  __shared__ half_t Bs[2][BN * LS];   // [n][k] (fragment-native)

  const int tid  = threadIdx.x;
  const int wave = tid >> 5, lane = tid & 31;
  const int wm = (wave & 3) * 32;    // wave row offset in tile (4 waves on M)
  const int wn = (wave >> 2) * 64;   // wave col offset in tile (2 waves on N)
  const int bm = blockIdx.y * BM;
  const int bn = blockIdx.x * BN;

  // staging maps: A 128x32 f32 (16/thread), B 128x32 halves (16/thread)
  const int arow = tid >> 1, acol = (tid & 1) * 16;
  const int brow = tid >> 1, bcol = (tid & 1) * 16;

  const float*  ag0 = A  + (size_t)(bm + arow) * K + acol;
  const half_t* bg0 = Bp + ((size_t)bn + brow) * 32 + bcol;   // + kb*N*32

  float  areg[16];
  half_t breg[16];
  const int nkb = K >> 5;   // BK = 32

  // ---- prologue: load + store tile 0 ----
  {
#pragma unroll
    for (int j = 0; j < 16; ++j) areg[j] = ag0[j];
#pragma unroll
    for (int j = 0; j < 16; ++j) breg[j] = bg0[j];
    half_t* ad = &As[0][arow * LS + acol];
#pragma unroll
    for (int j = 0; j < 16; ++j) ad[j] = (half_t)areg[j];
    half_t* bd = &Bs[0][brow * LS + bcol];
#pragma unroll
    for (int j = 0; j < 16; ++j) bd[j] = breg[j];
  }
  __syncthreads();

  v8f acc[2][4];
#pragma unroll
  for (int r = 0; r < 2; ++r)
#pragma unroll
    for (int c = 0; c < 4; ++c) acc[r][c] = (v8f){};

  const int g  = lane >> 4;
  const int fr = lane & 15;

  for (int kb = 0; kb < nkb; ++kb) {
    const int buf = kb & 1;
    if (kb + 1 < nkb) {               // issue next-tile global loads early
      const float*  ag = ag0 + (kb + 1) * BK;
      const half_t* bg = bg0 + (size_t)(kb + 1) * N * 32;
#pragma unroll
      for (int j = 0; j < 16; ++j) areg[j] = ag[j];
#pragma unroll
      for (int j = 0; j < 16; ++j) breg[j] = bg[j];
    }

    // fragments (ISA VGPR layouts; all chunks 16B-aligned in LDS)
    HFrag a[2], b[4];
#pragma unroll
    for (int r = 0; r < 2; ++r) {
      const half_t* ar = &As[buf][(wm + 16 * r + fr) * LS];
#pragma unroll
      for (int j = 0; j < 8; ++j) {
        a[r].h[j]     = ar[g * 8 + j];
        a[r].h[8 + j] = ar[16 + g * 8 + j];
      }
    }
#pragma unroll
    for (int c = 0; c < 4; ++c) {
      const half_t* br = &Bs[buf][(wn + 16 * c + fr) * LS];
#pragma unroll
      for (int j = 0; j < 8; ++j) {
        b[c].h[j]     = br[g * 8 + j];
        b[c].h[8 + j] = br[16 + g * 8 + j];
      }
    }

#pragma unroll
    for (int r = 0; r < 2; ++r)
#pragma unroll
      for (int c = 0; c < 4; ++c) acc[r][c] = wmma16(a[r], b[c], acc[r][c]);

    if (kb + 1 < nkb) {
      half_t* ad = &As[buf ^ 1][arow * LS + acol];
#pragma unroll
      for (int j = 0; j < 16; ++j) ad[j] = (half_t)areg[j];
      half_t* bd = &Bs[buf ^ 1][brow * LS + bcol];
#pragma unroll
      for (int j = 0; j < 16; ++j) bd[j] = breg[j];
      __syncthreads();
    }
  }

  // Epilogue. C layout per 16x16 tile: lane N = fr, VGPR r -> M = r + 8*g
  const int cn0 = bn + wn + fr;
  const int rb  = bm + wm + g * 8;
  float bv[4];
#pragma unroll
  for (int c = 0; c < 4; ++c) bv[c] = bias[cn0 + 16 * c];
#pragma unroll
  for (int r = 0; r < 2; ++r) {
    const size_t row0 = (size_t)(rb + 16 * r);
#pragma unroll
    for (int i = 0; i < 8; ++i) {
      const size_t row = row0 + i;
#pragma unroll
      for (int c = 0; c < 4; ++c) {
        float v = acc[r][c][i] + bv[c];
        if (do_gelu) v = gelu_exact(v);
        C[row * N + cn0 + 16 * c] = v;
      }
    }
  }
}

// -------- masking + shuffle + pos-embed + LayerNorm (one wave per token) ----
__global__ __launch_bounds__(256) void mask_pos_ln(
    const float* __restrict__ E,          // conv output [MTOKS, 768]
    const float* __restrict__ randomness, // [MTOKS, 3]
    const int*   __restrict__ perm,       // [MTOKS]
    const float* __restrict__ mask_tok,
    const float* __restrict__ temp_embd,  // [16,768]
    const float* __restrict__ spec_embd,  // [4,768]
    const float* __restrict__ gs, const float* __restrict__ gb,
    float* __restrict__ H)                // [MROWS, 768]
{
  const int wave = threadIdx.x >> 5, lane = threadIdx.x & 31;
  const int r = blockIdx.x * 8 + wave;            // 0..16383 exact
  const int b = r >> 6, tok = r & 63;
  const int tt = tok >> 2, ss = tok & 3;

  const float r0 = randomness[r * 3 + 0];
  const float r1 = randomness[r * 3 + 1];
  const float r2 = randomness[r * 3 + 2];
  const bool  sel = (r0 <= 0.2f);
  const float m  = (sel && (r1 <= 0.8f)) ? 1.0f : 0.0f;
  const float rn = (sel && (r1 > 0.8f) && (r2 <= 0.5f)) ? 1.0f : 0.0f;
  const float ke = 1.0f - m - rn;
  const int pe = perm[r];

  float x[24];
  float s = 0.0f;
#pragma unroll
  for (int j = 0; j < 24; ++j) {
    const int d = j * 32 + lane;
    const float e  = E[(size_t)r * D_MODEL + d];
    const float sh = E[(size_t)pe * D_MODEL + d];
    const float v = e * ke + mask_tok[d] * m + sh * rn
                  + temp_embd[tt * D_MODEL + d] + spec_embd[ss * D_MODEL + d];
    x[j] = v; s += v;
  }
  const float mean = wave_sum(s) * (1.0f / D_MODEL);
  float s2 = 0.0f;
#pragma unroll
  for (int j = 0; j < 24; ++j) { const float c = x[j] - mean; s2 += c * c; }
  const float rstd = rsqrtf(wave_sum(s2) * (1.0f / D_MODEL) + 1e-6f);

  float* out = H + ((size_t)(b * SEQ + 1 + tok)) * D_MODEL;
#pragma unroll
  for (int j = 0; j < 24; ++j) {
    const int d = j * 32 + lane;
    out[d] = (x[j] - mean) * rstd * gs[d] + gb[d];
  }
}

// -------------- dst = LN(X + R) (one wave per row, 768 wide) ---------------
__global__ __launch_bounds__(256) void residual_ln(
    const float* __restrict__ X, const float* __restrict__ R,
    const float* __restrict__ gs, const float* __restrict__ gb,
    float* __restrict__ dst, int nrows)
{
  const int wave = threadIdx.x >> 5, lane = threadIdx.x & 31;
  const int row = blockIdx.x * 8 + wave;
  if (row >= nrows) return;
  float x[24];
  float s = 0.0f;
#pragma unroll
  for (int j = 0; j < 24; ++j) {
    const int d = j * 32 + lane;
    const float v = X[(size_t)row * D_MODEL + d] + R[(size_t)row * D_MODEL + d];
    x[j] = v; s += v;
  }
  const float mean = wave_sum(s) * (1.0f / D_MODEL);
  float s2 = 0.0f;
#pragma unroll
  for (int j = 0; j < 24; ++j) { const float c = x[j] - mean; s2 += c * c; }
  const float rstd = rsqrtf(wave_sum(s2) * (1.0f / D_MODEL) + 1e-6f);
#pragma unroll
  for (int j = 0; j < 24; ++j) {
    const int d = j * 32 + lane;
    dst[(size_t)row * D_MODEL + d] = (x[j] - mean) * rstd * gs[d] + gb[d];
  }
}

// --------- attention: one wave per (b, head, q); S=65, Dkv=64 --------------
__global__ __launch_bounds__(256) void attention(
    const float* __restrict__ Q, const float* __restrict__ Km,
    const float* __restrict__ Vm, float* __restrict__ O)
{
  __shared__ float sc[8][72];
  const int wave = threadIdx.x >> 5, lane = threadIdx.x & 31;
  const int wid = blockIdx.x * 8 + wave;       // NB*NHEAD*SEQ = 199680 exact
  const int qi = wid % SEQ;
  const int t  = wid / SEQ;
  const int hh = t % NHEAD;
  const int b  = t / NHEAD;

  const size_t base = (size_t)b * SEQ;
  const float* qp = Q + (base + qi) * D_MODEL + hh * DKV;
  const float q0 = qp[lane], q1 = qp[lane + 32];

  for (int k = 0; k < SEQ; ++k) {
    const float* kp = Km + (base + k) * D_MODEL + hh * DKV;
    float s = q0 * kp[lane] + q1 * kp[lane + 32];
    s = wave_sum(s);
    if (lane == 0) sc[wave][k] = s * 0.125f;   // DKV^-0.5
  }
  const float a0 = sc[wave][lane];
  const float a1 = sc[wave][lane + 32];
  const float a2 = (lane == 0) ? sc[wave][64] : -3.0e38f;
  const float mx = wave_max(fmaxf(fmaxf(a0, a1), a2));

  const float e0 = __expf(a0 - mx);
  const float e1 = __expf(a1 - mx);
  const float e2 = (lane == 0) ? __expf(a2 - mx) : 0.0f;
  sc[wave][lane] = e0;
  sc[wave][lane + 32] = e1;
  if (lane == 0) sc[wave][64] = e2;
  const float inv = 1.0f / wave_sum(e0 + e1 + e2);

  float o0 = 0.0f, o1 = 0.0f;
  for (int k = 0; k < SEQ; ++k) {
    const float p = sc[wave][k];
    const float* vp = Vm + (base + k) * D_MODEL + hh * DKV;
    o0 += p * vp[lane];
    o1 += p * vp[lane + 32];
  }
  float* op = O + (base + qi) * D_MODEL + hh * DKV;
  op[lane]      = o0 * inv;
  op[lane + 32] = o1 * inv;
}

// ---------------------------------------------------------------------------
extern "C" void kernel_launch(void* const* d_in, const int* in_sizes, int n_in,
                              void* d_out, int out_size, void* d_ws, size_t ws_size,
                              hipStream_t stream) {
  (void)in_sizes; (void)n_in; (void)out_size; (void)ws_size;
  const float* inputs     = (const float*)d_in[0];
  const float* randomness = (const float*)d_in[1];
  const int*   perm       = (const int*)  d_in[2];
  const float* conv_w     = (const float*)d_in[3];
  const float* conv_b     = (const float*)d_in[4];
  const float* temp_embd  = (const float*)d_in[5];
  const float* spec_embd  = (const float*)d_in[6];
  const float* pos_ln_s   = (const float*)d_in[7];
  const float* pos_ln_b   = (const float*)d_in[8];
  const float* mask_tok   = (const float*)d_in[9];
  const float* agg_tok    = (const float*)d_in[10];
  const float* Wq = (const float*)d_in[11]; const float* bq = (const float*)d_in[12];
  const float* Wk = (const float*)d_in[13]; const float* bk = (const float*)d_in[14];
  const float* Wv = (const float*)d_in[15]; const float* bv = (const float*)d_in[16];
  const float* Wo = (const float*)d_in[17]; const float* bo = (const float*)d_in[18];
  const float* ln1_s = (const float*)d_in[19]; const float* ln1_b = (const float*)d_in[20];
  const float* W1 = (const float*)d_in[21]; const float* b1 = (const float*)d_in[22];
  const float* W2 = (const float*)d_in[23]; const float* b2 = (const float*)d_in[24];
  const float* ln2_s = (const float*)d_in[25]; const float* ln2_b = (const float*)d_in[26];

  // ---- workspace carve-up ----
  char* base = (char*)d_ws;
  size_t off = 0;
  auto carve = [&](size_t bytes) -> char* {
    char* p = base + off;
    off = (off + bytes + 255) & ~(size_t)255;
    return p;
  };
  const int SZ_QKV = D_MODEL * D_MODEL;       // 589824 per layer
  const int SZ_FF  = D_MODEL * D_FF;          // 2359296 per layer
  half_t* cw16 = (half_t*)carve((size_t)64 * D_MODEL * 2);
  half_t* wq16 = (half_t*)carve((size_t)2 * SZ_QKV * 2);
  half_t* wk16 = (half_t*)carve((size_t)2 * SZ_QKV * 2);
  half_t* wv16 = (half_t*)carve((size_t)2 * SZ_QKV * 2);
  half_t* wo16 = (half_t*)carve((size_t)2 * SZ_QKV * 2);
  half_t* w116 = (half_t*)carve((size_t)2 * SZ_FF * 2);
  half_t* w216 = (half_t*)carve((size_t)2 * SZ_FF * 2);
  float* P    = (float*)carve((size_t)MTOKS * 64 * 4);
  float* H    = (float*)carve((size_t)MROWS * D_MODEL * 4);
  float* Qb   = (float*)carve((size_t)MROWS * D_MODEL * 4);
  float* Kb   = (float*)carve((size_t)MROWS * D_MODEL * 4);
  float* Vb   = (float*)carve((size_t)MROWS * D_MODEL * 4);
  float* Ob   = (float*)carve((size_t)MROWS * D_MODEL * 4);
  float* T768 = (float*)carve((size_t)MROWS * D_MODEL * 4);
  float* TFF  = (float*)carve((size_t)MROWS * D_FF * 4);

  // ---- weight conversion f32 -> f16, fragment-native packing (per layer) ----
  cvt_pack_b<<<(64 * D_MODEL + 255) / 256, 256, 0, stream>>>(conv_w, cw16, 64, D_MODEL);
  for (int l = 0; l < 2; ++l) {
    const size_t oq = (size_t)l * SZ_QKV, of = (size_t)l * SZ_FF;
    cvt_pack_b<<<(SZ_QKV + 255) / 256, 256, 0, stream>>>(Wq + oq, wq16 + oq, D_MODEL, D_MODEL);
    cvt_pack_b<<<(SZ_QKV + 255) / 256, 256, 0, stream>>>(Wk + oq, wk16 + oq, D_MODEL, D_MODEL);
    cvt_pack_b<<<(SZ_QKV + 255) / 256, 256, 0, stream>>>(Wv + oq, wv16 + oq, D_MODEL, D_MODEL);
    cvt_pack_b<<<(SZ_QKV + 255) / 256, 256, 0, stream>>>(Wo + oq, wo16 + oq, D_MODEL, D_MODEL);
    cvt_pack_b<<<(SZ_FF + 255) / 256, 256, 0, stream>>>(W1 + of, w116 + of, D_MODEL, D_FF);
    cvt_pack_b<<<(SZ_FF + 255) / 256, 256, 0, stream>>>(W2 + of, w216 + of, D_FF, D_MODEL);
  }

  // ---- patchify + conv GEMM (e_raw -> first 16384 rows of T768) ----
  patchify<<<MTOKS * 64 / 256, 256, 0, stream>>>(inputs, P);
  gemm_wmma<<<dim3(D_MODEL / BN, MTOKS / BM), 256, 0, stream>>>(
      P, cw16, T768, conv_b, MTOKS, D_MODEL, 64, 0);

  // ---- AGG rows + masked/pos/LN token rows into H ----
  set_agg<<<NB * D_MODEL / 256, 256, 0, stream>>>(agg_tok, H);
  mask_pos_ln<<<MTOKS / 8, 256, 0, stream>>>(T768, randomness, perm, mask_tok,
                                             temp_embd, spec_embd, pos_ln_s,
                                             pos_ln_b, H);

  // ---- transformer layers ----
  const dim3 g768(D_MODEL / BN, MROWS / BM);   // 6 x 130
  const dim3 gFF (D_FF / BN,   MROWS / BM);    // 24 x 130
  for (int l = 0; l < 2; ++l) {
    gemm_wmma<<<g768, 256, 0, stream>>>(H, wq16 + (size_t)l * SZ_QKV, Qb,
                                        bq + l * D_MODEL, MROWS, D_MODEL, D_MODEL, 0);
    gemm_wmma<<<g768, 256, 0, stream>>>(H, wk16 + (size_t)l * SZ_QKV, Kb,
                                        bk + l * D_MODEL, MROWS, D_MODEL, D_MODEL, 0);
    gemm_wmma<<<g768, 256, 0, stream>>>(H, wv16 + (size_t)l * SZ_QKV, Vb,
                                        bv + l * D_MODEL, MROWS, D_MODEL, D_MODEL, 0);
    attention<<<NB * NHEAD * SEQ / 8, 256, 0, stream>>>(Qb, Kb, Vb, Ob);
    gemm_wmma<<<g768, 256, 0, stream>>>(Ob, wo16 + (size_t)l * SZ_QKV, T768,
                                        bo + l * D_MODEL, MROWS, D_MODEL, D_MODEL, 0);
    residual_ln<<<MROWS / 8, 256, 0, stream>>>(H, T768, ln1_s + l * D_MODEL,
                                               ln1_b + l * D_MODEL, H, MROWS);
    gemm_wmma<<<gFF, 256, 0, stream>>>(H, w116 + (size_t)l * SZ_FF, TFF,
                                       b1 + l * D_FF, MROWS, D_FF, D_MODEL, 1);
    gemm_wmma<<<g768, 256, 0, stream>>>(TFF, w216 + (size_t)l * SZ_FF, T768,
                                        b2 + l * D_MODEL, MROWS, D_MODEL, D_FF, 0);
    float* dst = (l == 1) ? (float*)d_out : H;
    residual_ln<<<MROWS / 8, 256, 0, stream>>>(H, T768, ln2_s + l * D_MODEL,
                                               ln2_b + l * D_MODEL, dst, MROWS);
  }
}
